// ScaledDotProductAttention_54889682043185
// MI455X (gfx1250) — compile-verified, matrix operand order checked by reading, not run on hardware
//
#include <hip/hip_runtime.h>
#include <hip/hip_bf16.h>
#include <math.h>

typedef __attribute__((ext_vector_type(16))) __bf16 v16bf;
typedef __attribute__((ext_vector_type(8)))  float  v8f;

#define SLEN  2048
#define DK    64
#define NWAVE 4
#define KCHUNK (SLEN / NWAVE)   // 512

// Element e of a 16-bit WMMA A/B fragment maps to K-index (per CDNA5 ISA 7.12.2):
// lanes 0-15 (g=0): VGPR0 holds K=0,1 ... VGPR3 K=6,7; VGPR4-7 K=16..23
// lanes 16-31 (g=1): K=8..15 and K=24..31
__device__ __forceinline__ int kmap(int e, int g) {
    return ((e >> 3) << 4) + (g << 3) + (e & 7);
}

// Load 16 fp32 values from a row-major row (global or LDS) into a bf16 fragment:
// element e -> rowp[kbase + kmap(e,g)]. Uses 4x 128-bit loads.
__device__ __forceinline__ v16bf load_row_frag(const float* __restrict__ rowp, int kbase, int g) {
    v16bf f;
    const float* p = rowp + kbase + (g << 3);
#pragma unroll
    for (int h = 0; h < 2; ++h) {
        float4 lo = *(const float4*)(p + (h << 4));
        float4 hi = *(const float4*)(p + (h << 4) + 4);
        f[h * 8 + 0] = (__bf16)lo.x; f[h * 8 + 1] = (__bf16)lo.y;
        f[h * 8 + 2] = (__bf16)lo.z; f[h * 8 + 3] = (__bf16)lo.w;
        f[h * 8 + 4] = (__bf16)hi.x; f[h * 8 + 5] = (__bf16)hi.y;
        f[h * 8 + 6] = (__bf16)hi.z; f[h * 8 + 7] = (__bf16)hi.w;
    }
    return f;
}

__global__ void sdpa_bf16wmma_kernel(const float* __restrict__ Q,
                                     const float* __restrict__ K,
                                     const float* __restrict__ V,
                                     const float* __restrict__ mask,
                                     float* __restrict__ ctx_out,
                                     float* __restrict__ attn_out,
                                     int S, int H) {
    extern __shared__ float smem[];
    float* sc     = smem;             // [16][S] exp-scores (normalized in place)
    float* rowsum = smem + 16 * S;    // [16]
    float* cacc   = rowsum + 16;      // [16][DK] context accumulator

    const int tid  = threadIdx.x;
    const int lane = tid & 31;
    const int wid  = tid >> 5;        // 4 waves per block
    const int g    = lane >> 4;       // lane group (0: lanes 0-15, 1: lanes 16-31)
    const int n    = lane & 15;

    const int qtiles = S >> 4;
    const int bh = blockIdx.x / qtiles;
    const int q0 = (blockIdx.x % qtiles) << 4;
    const int b  = bh / H;

    const float scale = 0.125f;       // 1/sqrt(64)

    if (tid < 16) rowsum[tid] = 0.0f;
    for (int i = tid; i < 16 * DK; i += blockDim.x) cacc[i] = 0.0f;
    __syncthreads();

    const float* Qb = Q + (size_t)(bh * S + q0) * DK;
    const float* Kb = K + (size_t)bh * S * DK;
    const float* Vb = V + (size_t)bh * S * DK;
    const float* Mb = mask + (size_t)b * S * S + (size_t)q0 * S;

    // A-fragments of Q (row = n, K-dim = d split 0..31 / 32..63)
    const v16bf aq0 = load_row_frag(Qb + n * DK, 0, g);
    const v16bf aq1 = load_row_frag(Qb + n * DK, 32, g);

    const int wk0 = wid * KCHUNK;

    float ps[8];
#pragma unroll
    for (int r = 0; r < 8; ++r) ps[r] = 0.0f;

    // ---- Pass 1: scores = exp(QK^T * scale) * mask, kept unnormalized in LDS ----
    for (int kt = 0; kt < KCHUNK / 16; ++kt) {
        const int kr = wk0 + kt * 16;
        // B-fragment of K^T: B[d][kcol] = K[kr+n][d]
        const float* krow = Kb + (size_t)(kr + n) * DK;
        v16bf bk0 = load_row_frag(krow, 0, g);
        v16bf bk1 = load_row_frag(krow, 32, g);
        if (kt + 1 < KCHUNK / 16)
            __builtin_prefetch(Kb + (size_t)(kr + 16 + n) * DK, 0, 0);

        v8f c = {};
        c = __builtin_amdgcn_wmma_f32_16x16x32_bf16(false, aq0, false, bk0, (short)0, c, false, false);
        c = __builtin_amdgcn_wmma_f32_16x16x32_bf16(false, aq1, false, bk1, (short)0, c, false, false);

#pragma unroll
        for (int r = 0; r < 8; ++r) {
            const int m   = r + (g << 3);       // C/D layout: M = vgpr + 8*lanegroup
            const int col = kr + n;             // N = lane & 15
            float v = __expf(c[r] * scale) * Mb[(size_t)m * S + col];
            sc[m * S + col] = v;
            ps[r] += v;
        }
    }
#pragma unroll
    for (int r = 0; r < 8; ++r) atomicAdd(&rowsum[r + (g << 3)], ps[r]);
    __syncthreads();

    // ---- Pass 2a: normalize this wave's 16 x 512 chunk in place in LDS ----
    for (int m = 0; m < 16; ++m) {
        const float inv = 1.0f / (rowsum[m] + 1e-8f);
        float* row = sc + m * S + wk0;
        for (int c4 = lane * 4; c4 < KCHUNK; c4 += 32 * 4) {
            float4 v = *(const float4*)(row + c4);
            v.x *= inv; v.y *= inv; v.z *= inv; v.w *= inv;
            *(float4*)(row + c4) = v;
        }
    }
    // Make our LDS writes visible before the async engine reads them.
    asm volatile("s_wait_dscnt 0x0" ::: "memory");

    // ---- Pass 2b: drain attn to HBM via async LDS->global stores (ASYNCcnt).
    // Each b128 issue moves 512B/wave; overlaps with the PV WMMA loop below. ----
    for (int m = 0; m < 16; ++m) {
        const unsigned long long gbase =
            (unsigned long long)(uintptr_t)(attn_out + (size_t)(bh * S + q0 + m) * S + wk0);
        const unsigned lbase = (unsigned)(uintptr_t)(sc + m * S + wk0);
#pragma unroll
        for (int it = 0; it < KCHUNK / (32 * 4); ++it) {   // 4 issues per row
            unsigned long long gaddr = gbase + (unsigned)(it * 512 + lane * 16);
            unsigned           laddr = lbase + (unsigned)(it * 512 + lane * 16);
            asm volatile("global_store_async_from_lds_b128 %0, %1, off"
                         :
                         : "v"(gaddr), "v"(laddr)
                         : "memory");
        }
    }

    // ---- Pass 3: context partial = P(16 x KCHUNK) @ V(KCHUNK x 64)
    //      (sc is already normalized, so A-fragments load straight via ds_load_b128) ----
    v8f acc[4] = {};
    for (int kt = 0; kt < KCHUNK / 32; ++kt) {
        const int kk0 = wk0 + kt * 32;
        v16bf ap = load_row_frag(sc + n * S, kk0, g);   // row n of normalized P
#pragma unroll
        for (int d = 0; d < 4; ++d) {
            v16bf bv;   // B[kk][dcol] = V[kk0+kmap(e)][d*16 + n]
#pragma unroll
            for (int e = 0; e < 16; ++e)
                bv[e] = (__bf16)Vb[(size_t)(kk0 + kmap(e, g)) * DK + d * 16 + n];
            acc[d] = __builtin_amdgcn_wmma_f32_16x16x32_bf16(false, ap, false, bv, (short)0, acc[d], false, false);
        }
    }
    // reduce 4 wave-partials of the 16x64 context tile in LDS
#pragma unroll
    for (int d = 0; d < 4; ++d)
#pragma unroll
        for (int r = 0; r < 8; ++r)
            atomicAdd(&cacc[(r + (g << 3)) * DK + d * 16 + n], acc[d][r]);
    __syncthreads();

    for (int i = tid; i < 16 * DK; i += blockDim.x) {
        const int m = i >> 6, d = i & 63;
        ctx_out[(size_t)(bh * S + q0 + m) * DK + d] = cacc[i];
    }

    // Ensure all async attn stores have completed before the wave retires.
    asm volatile("s_wait_asynccnt 0x0" ::: "memory");
}

extern "C" void kernel_launch(void* const* d_in, const int* in_sizes, int n_in,
                              void* d_out, int out_size, void* d_ws, size_t ws_size,
                              hipStream_t stream) {
    (void)in_sizes; (void)n_in; (void)out_size; (void)d_ws; (void)ws_size;
    const float* Q = (const float*)d_in[0];
    const float* K = (const float*)d_in[1];
    const float* V = (const float*)d_in[2];
    const float* M = (const float*)d_in[3];

    const int B = 4, H = 16, S = SLEN;
    float* ctx  = (float*)d_out;                       // (B,H,S,DK) first
    float* attn = ctx + (size_t)B * H * S * DK;        // then (B,H,S,S)

    const size_t shmem = (size_t)(16 * S + 16 + 16 * DK) * sizeof(float); // ~132 KB
    hipFuncSetAttribute((const void*)sdpa_bf16wmma_kernel,
                        hipFuncAttributeMaxDynamicSharedMemorySize, (int)shmem);

    dim3 grid(B * H * (S / 16));   // 8192 blocks
    dim3 block(128);               // 4 waves (wave32)
    sdpa_bf16wmma_kernel<<<grid, block, shmem, stream>>>(Q, K, V, M, ctx, attn, S, H);
}
